// TransformerBlock_42228118454556
// MI455X (gfx1250) — matrix operand order repaired and checked by reading, not compiled, not run on hardware
//
#include <hip/hip_runtime.h>
#include <hip/hip_fp16.h>

// ---------------------------------------------------------------------------
// Transformer block for MI455X (gfx1250, wave32, WMMA).
// GEMMs: V_WMMA_F32_16X16X32_F16 (f16 in, f32 acc), LDS double-buffered with
// CDNA5 async global->LDS copies (ASYNCcnt). Attention: flash-style with
// window+global skip logic, fragments loaded straight from global memory.
// ---------------------------------------------------------------------------

#define EMB   1024
#define NH    16
#define HD    64
#define TSEQ  2048
#define BATCH 2
#define NROWS (BATCH * TSEQ)   // 4096 token rows
#define WINDOW 256

// Flip to 0 if the async-to-LDS asm ever fails to assemble.
#define USE_ASYNC_LDS 1

typedef __attribute__((ext_vector_type(16))) _Float16 v16h;
typedef __attribute__((ext_vector_type(8)))  _Float16 h8;
typedef __attribute__((ext_vector_type(8)))  float    v8f;

__device__ __forceinline__ h8 ld8(const _Float16* p) { return *(const h8*)p; }

__device__ __forceinline__ v16h mkfrag(h8 lo, h8 hi) {
  union { v16h v; h8 h[2]; } u;
  u.h[0] = lo; u.h[1] = hi;
  return u.v;
}

#if USE_ASYNC_LDS
// CDNA5 async copy: 16B global -> LDS, no VGPR staging, tracked by ASYNCcnt.
// VDST carries the LDS byte address (low 32 bits of the generic pointer per
// the LDS aperture rule, ISA 10.2); VADDR is the 64-bit global address (GV).
__device__ __forceinline__ void async_copy16(const _Float16* g, _Float16* s) {
  unsigned lds = (unsigned)(uintptr_t)s;
  asm volatile("global_load_async_to_lds_b128 %0, %1, off"
               :: "v"(lds), "v"(g) : "memory");
}
__device__ __forceinline__ void wait_async0() {
  asm volatile("s_wait_asynccnt 0x0" ::: "memory");
}
#endif

// ---------------------------------------------------------------------------
// Weight convert + transpose: W[K][N] f32  ->  Wt[N][K] f16
// ---------------------------------------------------------------------------
__global__ __launch_bounds__(256) void convT_kernel(const float* __restrict__ W,
                                                    _Float16* __restrict__ Wt,
                                                    int Kd, int Nd) {
  __shared__ float tile[32][33];
  const int n0 = blockIdx.x * 32, k0 = blockIdx.y * 32;
  const int tx = threadIdx.x, ty = threadIdx.y;   // block (32,8)
#pragma unroll
  for (int i = 0; i < 32; i += 8)
    tile[ty + i][tx] = W[(size_t)(k0 + ty + i) * Nd + (n0 + tx)];
  __syncthreads();
#pragma unroll
  for (int i = 0; i < 32; i += 8)
    Wt[(size_t)(n0 + ty + i) * Kd + (k0 + tx)] = (_Float16)tile[tx][ty + i];
}

// ---------------------------------------------------------------------------
// LayerNorm (row = 1024 floats) -> f16
// ---------------------------------------------------------------------------
__global__ __launch_bounds__(256) void ln_kernel(const float* __restrict__ x,
                                                 const float* __restrict__ g,
                                                 const float* __restrict__ b,
                                                 _Float16* __restrict__ out) {
  __shared__ float red[16];
  const int row = blockIdx.x;
  const int tid = threadIdx.x;
  const float* xr = x + (size_t)row * EMB;
  float4 v = ((const float4*)xr)[tid];
  float s  = v.x + v.y + v.z + v.w;
  float s2 = v.x * v.x + v.y * v.y + v.z * v.z + v.w * v.w;
#pragma unroll
  for (int m = 1; m < 32; m <<= 1) { s += __shfl_xor(s, m); s2 += __shfl_xor(s2, m); }
  const int wave = tid >> 5, lane = tid & 31;
  if (lane == 0) { red[wave] = s; red[8 + wave] = s2; }
  __syncthreads();
  float ts = 0.f, ts2 = 0.f;
#pragma unroll
  for (int w = 0; w < 8; w++) { ts += red[w]; ts2 += red[8 + w]; }
  const float mu   = ts * (1.0f / EMB);
  const float var  = ts2 * (1.0f / EMB) - mu * mu;
  const float rstd = rsqrtf(var + 1e-5f);
  float4 gg = ((const float4*)g)[tid];
  float4 bb = ((const float4*)b)[tid];
  _Float16* o = out + (size_t)row * EMB + tid * 4;
  o[0] = (_Float16)((v.x - mu) * rstd * gg.x + bb.x);
  o[1] = (_Float16)((v.y - mu) * rstd * gg.y + bb.y);
  o[2] = (_Float16)((v.z - mu) * rstd * gg.z + bb.z);
  o[3] = (_Float16)((v.w - mu) * rstd * gg.w + bb.w);
}

// ---------------------------------------------------------------------------
// Generic WMMA GEMM:  C[M][N] = A[M][K] (f16) x Wt[N][K]^T (f16) + bias
//   mode 0: store f16 row-major           (outH)
//   mode 1: store f16 V-transposed        (outH[((b*NH+h)*HD+d)*T + t])
//   mode 2: exact-GELU then store f16     (outH)
//   mode 3: outF = res + alpha * C        (f32)
// 128x128 block tile, 8 waves of 32x64, K sliced by 32, double-buffered LDS
// filled with async global->LDS copies.
// ---------------------------------------------------------------------------
__global__ __launch_bounds__(256) void gemm_kernel(
    const _Float16* __restrict__ A, const _Float16* __restrict__ Bt,
    const float* __restrict__ bias, int M, int N, int K, int mode,
    const float* __restrict__ res, const float* __restrict__ alphaPtr,
    float* __restrict__ outF, _Float16* __restrict__ outH) {
  (void)M;
  __shared__ _Float16 sA[2][128][40];   // +8 pad: 80B row stride, 16B aligned
  __shared__ _Float16 sB[2][128][40];

  const int tid  = threadIdx.x;
  const int lane = tid & 31;
  const int wave = tid >> 5;
  const int l    = lane & 15;
  const int hf   = lane >> 4;
  const int wm   = (wave >> 1) * 32;    // wave row offset in 128
  const int wn   = (wave & 1) * 64;     // wave col offset in 128
  const size_t rowBase = (size_t)blockIdx.y * 128;
  const size_t colBase = (size_t)blockIdx.x * 128;

  const int lrow = tid >> 1;            // cooperative loader: 128 rows x 32 k
  const int lkp  = (tid & 1) * 16;
  const _Float16* Ag = A  + (rowBase + lrow) * (size_t)K + lkp;
  const _Float16* Bg = Bt + (colBase + lrow) * (size_t)K + lkp;

  v8f acc[2][4] = {};

  auto loadTile = [&](int buf, int kk) {
#if USE_ASYNC_LDS
    async_copy16(Ag + kk,     &sA[buf][lrow][lkp]);
    async_copy16(Ag + kk + 8, &sA[buf][lrow][lkp + 8]);
    async_copy16(Bg + kk,     &sB[buf][lrow][lkp]);
    async_copy16(Bg + kk + 8, &sB[buf][lrow][lkp + 8]);
#else
    uint4 a0 = *(const uint4*)(Ag + kk);
    uint4 a1 = *(const uint4*)(Ag + kk + 8);
    uint4 b0 = *(const uint4*)(Bg + kk);
    uint4 b1 = *(const uint4*)(Bg + kk + 8);
    *(uint4*)&sA[buf][lrow][lkp]     = a0;
    *(uint4*)&sA[buf][lrow][lkp + 8] = a1;
    *(uint4*)&sB[buf][lrow][lkp]     = b0;
    *(uint4*)&sB[buf][lrow][lkp + 8] = b1;
#endif
  };

  const int nK = K >> 5;
  loadTile(0, 0);
#if USE_ASYNC_LDS
  wait_async0();
#endif
  __syncthreads();
  for (int it = 0; it < nK; ++it) {
    if (it + 1 < nK) loadTile((it + 1) & 1, (it + 1) << 5);
    const int bsel = it & 1;
    v16h af[2], bf[4];
#pragma unroll
    for (int i = 0; i < 2; i++) {
      const _Float16* p = &sA[bsel][wm + i * 16 + l][0];
      // A 16x32 layout: lanes<16 K{0-7,16-23}, lanes>=16 K{8-15,24-31}
      af[i] = mkfrag(ld8(p + 8 * hf), ld8(p + 16 + 8 * hf));
    }
#pragma unroll
    for (int j = 0; j < 4; j++) {
      const _Float16* p = &sB[bsel][wn + j * 16 + l][0];
      // B 32x16 layout: lanes<16 K{0-7,16-23}, lanes>=16 K{8-15,24-31}
      bf[j] = mkfrag(ld8(p + 8 * hf), ld8(p + 16 + 8 * hf));
    }
#pragma unroll
    for (int i = 0; i < 2; i++)
#pragma unroll
      for (int j = 0; j < 4; j++)
        acc[i][j] = __builtin_amdgcn_wmma_f32_16x16x32_f16(
            false, af[i], false, bf[j], (short)0, acc[i][j], false, false);
#if USE_ASYNC_LDS
    wait_async0();            // next buffer resident before the barrier
#endif
    __syncthreads();
  }

  const float alpha = alphaPtr ? alphaPtr[0] : 1.0f;
#pragma unroll
  for (int i = 0; i < 2; i++) {
#pragma unroll
    for (int j = 0; j < 4; j++) {
#pragma unroll
      for (int r = 0; r < 8; r++) {
        const size_t row = rowBase + wm + i * 16 + r + 8 * hf;  // C: VGPR r -> M=r (+8 hi half)
        const size_t col = colBase + wn + j * 16 + l;           // C: lane -> N
        float c = acc[i][j][r] + bias[col];
        if (mode == 0) {
          outH[row * (size_t)N + col] = (_Float16)c;
        } else if (mode == 1) {
          const size_t bb = row >> 11, t = row & (TSEQ - 1);
          const size_t h = col >> 6, d = col & (HD - 1);
          outH[((bb * NH + h) * HD + d) * (size_t)TSEQ + t] = (_Float16)c;
        } else if (mode == 2) {
          float ge = 0.5f * c * (1.0f + erff(c * 0.70710678118654752f));
          outH[row * (size_t)N + col] = (_Float16)ge;
        } else {
          const size_t idx = row * (size_t)N + col;
          outF[idx] = res[idx] + alpha * c;
        }
      }
    }
  }
}

// ---------------------------------------------------------------------------
// Flash attention, one wave per 16-query tile. Window+global mask, skip
// chunks fully outside the window with no global rows/cols.
// Q,K: [b*T + t][h*64 + d] f16.  Vt: [((b*NH+h)*HD + d)][t] f16.
// ---------------------------------------------------------------------------
__global__ __launch_bounds__(256) void attn_kernel(
    const _Float16* __restrict__ Q, const _Float16* __restrict__ Kf,
    const _Float16* __restrict__ Vt, const unsigned char* __restrict__ gm,
    _Float16* __restrict__ Aout) {
  __shared__ _Float16 sP[8][16 * 32];   // per-wave P staging (1KB)
  const int tid  = threadIdx.x;
  const int lane = tid & 31;
  const int wave = tid >> 5;
  const int l    = lane & 15;
  const int hf   = lane >> 4;
  const int wid  = blockIdx.x * 8 + wave;      // 0 .. B*NH*(T/16)-1 = 4095
  const int bb   = wid >> 11;                  // / (NH * T/16)
  const int h    = (wid >> 7) & (NH - 1);
  const int qt   = wid & ((TSEQ / 16) - 1);
  const int q0   = qt * 16;
  _Float16* sPw  = &sP[wave][0];

  // Q fragments (A 16x32 x2, covering d = 0..63); lane row = q0 + l
  const _Float16* qbase = Q + ((size_t)(bb * TSEQ + q0 + l)) * EMB + h * HD;
  v16h qf[2];
#pragma unroll
  for (int s = 0; s < 2; s++)
    qf[s] = mkfrag(ld8(qbase + s * 32 + 8 * hf),
                   ld8(qbase + s * 32 + 16 + 8 * hf));

  int rg[8];
#pragma unroll
  for (int r = 0; r < 8; r++) rg[r] = gm[q0 + r + 8 * hf];
  int anyRowG = 0;
#pragma unroll
  for (int i = 0; i < 16; i++) anyRowG |= gm[q0 + i];

  float mrow[8], lrow[8];
  v8f o[4] = {};
#pragma unroll
  for (int r = 0; r < 8; r++) { mrow[r] = -1e30f; lrow[r] = 0.0f; }

  const int nch = ((q0 + 15) >> 5) + 1;
  for (int c = 0; c < nch; c++) {
    const int c0 = c << 5;
    int cg = gm[c0 + lane];
    int anyColG = cg;
#pragma unroll
    for (int m = 1; m < 32; m <<= 1) anyColG |= __shfl_xor(anyColG, m);
    if ((c0 + 31 < q0 - (WINDOW - 1)) && !anyRowG && !anyColG) continue;

    // scores for two 16-key tiles
    v8f S[2];
#pragma unroll
    for (int t = 0; t < 2; t++) {
      const int ct = c0 + t * 16;
      const _Float16* kb = Kf + ((size_t)(bb * TSEQ + ct + l)) * EMB + h * HD;
      v8f sc = {};
#pragma unroll
      for (int s = 0; s < 2; s++) {
        v16h bfK = mkfrag(ld8(kb + s * 32 + 8 * hf),
                          ld8(kb + s * 32 + 16 + 8 * hf));
        sc = __builtin_amdgcn_wmma_f32_16x16x32_f16(
            false, qf[s], false, bfK, (short)0, sc, false, false);
      }
      S[t] = sc;
    }

    // mask + scale (1/sqrt(64) = 0.125)
    const int cg0 = gm[c0 + l], cg1 = gm[c0 + 16 + l];
#pragma unroll
    for (int r = 0; r < 8; r++) {
      const int row = q0 + r + 8 * hf;
      const int col0 = c0 + l, col1 = c0 + 16 + l;
      const bool ok0 = (col0 <= row) && ((row - col0 < WINDOW) || rg[r] || cg0);
      const bool ok1 = (col1 <= row) && ((row - col1 < WINDOW) || rg[r] || cg1);
      S[0][r] = ok0 ? S[0][r] * 0.125f : -1e30f;
      S[1][r] = ok1 ? S[1][r] * 0.125f : -1e30f;
    }

    // online softmax; rows live at (VGPR r, lane-half), reduce over 16 lanes
    float alpha[8];
#pragma unroll
    for (int r = 0; r < 8; r++) {
      float mx = fmaxf(S[0][r], S[1][r]);
#pragma unroll
      for (int m = 1; m < 16; m <<= 1) mx = fmaxf(mx, __shfl_xor(mx, m));
      const float mnew = fmaxf(mrow[r], mx);
      alpha[r] = __expf(mrow[r] - mnew);
      const float p0 = __expf(S[0][r] - mnew);
      const float p1 = __expf(S[1][r] - mnew);
      S[0][r] = p0; S[1][r] = p1;
      float ps = p0 + p1;
#pragma unroll
      for (int m = 1; m < 16; m <<= 1) ps += __shfl_xor(ps, m);
      lrow[r] = alpha[r] * lrow[r] + ps;
      mrow[r] = mnew;
    }
#pragma unroll
    for (int j = 0; j < 4; j++)
#pragma unroll
      for (int r = 0; r < 8; r++) o[j][r] *= alpha[r];

    // stage P (16x32, C layout -> row-major f16 in LDS)
#pragma unroll
    for (int r = 0; r < 8; r++) {
      sPw[(r + 8 * hf) * 32 + l]      = (_Float16)S[0][r];
      sPw[(r + 8 * hf) * 32 + 16 + l] = (_Float16)S[1][r];
    }
    asm volatile("s_wait_dscnt 0" ::: "memory");
    v16h pf = mkfrag(ld8(sPw + l * 32 + 8 * hf),
                     ld8(sPw + l * 32 + 16 + 8 * hf));
    // P (16x32) x V (32x64) as 4 WMMAs over dim tiles
#pragma unroll
    for (int j = 0; j < 4; j++) {
      const _Float16* vb =
          Vt + ((size_t)((bb * NH + h) * HD + j * 16 + l)) * TSEQ + c0;
      v16h vf = mkfrag(ld8(vb + 8 * hf), ld8(vb + 16 + 8 * hf));
      o[j] = __builtin_amdgcn_wmma_f32_16x16x32_f16(
          false, pf, false, vf, (short)0, o[j], false, false);
    }
  }

#pragma unroll
  for (int r = 0; r < 8; r++) lrow[r] = 1.0f / lrow[r];
#pragma unroll
  for (int j = 0; j < 4; j++)
#pragma unroll
    for (int r = 0; r < 8; r++) {
      const size_t row = (size_t)bb * TSEQ + q0 + r + 8 * hf;
      Aout[row * EMB + h * HD + j * 16 + l] = (_Float16)(o[j][r] * lrow[r]);
    }
}

// ---------------------------------------------------------------------------
// Host orchestration (graph-capture safe: only kernel launches on stream)
// ---------------------------------------------------------------------------
extern "C" void kernel_launch(void* const* d_in, const int* in_sizes, int n_in,
                              void* d_out, int out_size, void* d_ws,
                              size_t ws_size, hipStream_t stream) {
  (void)in_sizes; (void)n_in; (void)out_size; (void)ws_size;
  const float* x  = (const float*)d_in[0];
  const unsigned char* gm = (const unsigned char*)d_in[1];  // jnp bool -> 1 byte
  const float* Wq = (const float*)d_in[2];  const float* bq  = (const float*)d_in[3];
  const float* Wk = (const float*)d_in[4];  const float* bk  = (const float*)d_in[5];
  const float* Wv = (const float*)d_in[6];  const float* bv  = (const float*)d_in[7];
  const float* Wo = (const float*)d_in[8];  const float* bo  = (const float*)d_in[9];
  const float* ln1g = (const float*)d_in[10]; const float* ln1b = (const float*)d_in[11];
  const float* ln2g = (const float*)d_in[12]; const float* ln2b = (const float*)d_in[13];
  const float* W1 = (const float*)d_in[14]; const float* bf1 = (const float*)d_in[15];
  const float* W2 = (const float*)d_in[16]; const float* bf2 = (const float*)d_in[17];
  const float* aAttn = (const float*)d_in[18];
  const float* aFF   = (const float*)d_in[19];

  char* ws = (char*)d_ws;
  size_t off = 0;
  auto alloc = [&](size_t bytes) {
    size_t r = off; off += (bytes + 255) & ~(size_t)255; return r;
  };
  _Float16* xh    = (_Float16*)(ws + alloc((size_t)NROWS * EMB * 2));
  _Float16* qh    = (_Float16*)(ws + alloc((size_t)NROWS * EMB * 2));
  _Float16* kh    = (_Float16*)(ws + alloc((size_t)NROWS * EMB * 2));
  _Float16* vth   = (_Float16*)(ws + alloc((size_t)NROWS * EMB * 2));
  _Float16* atth  = (_Float16*)(ws + alloc((size_t)NROWS * EMB * 2));
  _Float16* x2h   = (_Float16*)(ws + alloc((size_t)NROWS * EMB * 2));
  float*    x1    = (float*)   (ws + alloc((size_t)NROWS * EMB * 4));
  _Float16* hff   = (_Float16*)(ws + alloc((size_t)NROWS * 4 * EMB * 2));
  _Float16* wqT   = (_Float16*)(ws + alloc((size_t)EMB * EMB * 2));
  _Float16* wkT   = (_Float16*)(ws + alloc((size_t)EMB * EMB * 2));
  _Float16* wvT   = (_Float16*)(ws + alloc((size_t)EMB * EMB * 2));
  _Float16* woT   = (_Float16*)(ws + alloc((size_t)EMB * EMB * 2));
  _Float16* w1T   = (_Float16*)(ws + alloc((size_t)EMB * 4 * EMB * 2));
  _Float16* w2T   = (_Float16*)(ws + alloc((size_t)4 * EMB * EMB * 2));

  const dim3 tb(32, 8);
  convT_kernel<<<dim3(EMB / 32, EMB / 32), tb, 0, stream>>>(Wq, wqT, EMB, EMB);
  convT_kernel<<<dim3(EMB / 32, EMB / 32), tb, 0, stream>>>(Wk, wkT, EMB, EMB);
  convT_kernel<<<dim3(EMB / 32, EMB / 32), tb, 0, stream>>>(Wv, wvT, EMB, EMB);
  convT_kernel<<<dim3(EMB / 32, EMB / 32), tb, 0, stream>>>(Wo, woT, EMB, EMB);
  convT_kernel<<<dim3(4 * EMB / 32, EMB / 32), tb, 0, stream>>>(W1, w1T, EMB, 4 * EMB);
  convT_kernel<<<dim3(EMB / 32, 4 * EMB / 32), tb, 0, stream>>>(W2, w2T, 4 * EMB, EMB);

  ln_kernel<<<NROWS, 256, 0, stream>>>(x, ln1g, ln1b, xh);

  // QKV projections
  gemm_kernel<<<dim3(EMB / 128, NROWS / 128), 256, 0, stream>>>(
      xh, wqT, bq, NROWS, EMB, EMB, 0, nullptr, nullptr, nullptr, qh);
  gemm_kernel<<<dim3(EMB / 128, NROWS / 128), 256, 0, stream>>>(
      xh, wkT, bk, NROWS, EMB, EMB, 0, nullptr, nullptr, nullptr, kh);
  gemm_kernel<<<dim3(EMB / 128, NROWS / 128), 256, 0, stream>>>(
      xh, wvT, bv, NROWS, EMB, EMB, 1, nullptr, nullptr, nullptr, vth);

  // attention: B*NH*(T/16) = 4096 wave tiles / 8 waves per block
  attn_kernel<<<(BATCH * NH * (TSEQ / 16)) / 8, 256, 0, stream>>>(
      qh, kh, vth, gm, atth);

  // O projection + residual (x1 = x + alpha_attn * attn@Wo + bo)
  gemm_kernel<<<dim3(EMB / 128, NROWS / 128), 256, 0, stream>>>(
      atth, woT, bo, NROWS, EMB, EMB, 3, x, aAttn, x1, nullptr);

  ln_kernel<<<NROWS, 256, 0, stream>>>(x1, ln2g, ln2b, x2h);

  // FF up + GELU
  gemm_kernel<<<dim3(4 * EMB / 128, NROWS / 128), 256, 0, stream>>>(
      x2h, w1T, bf1, NROWS, 4 * EMB, EMB, 2, nullptr, nullptr, nullptr, hff);

  // FF down + residual -> final output (f32)
  gemm_kernel<<<dim3(EMB / 128, NROWS / 128), 256, 0, stream>>>(
      hff, w2T, bf2, NROWS, EMB, 4 * EMB, 3, x1, aFF, (float*)d_out, nullptr);
}
